// RSE_Mamba2_63677185130932
// MI455X (gfx1250) — matrix operand order repaired and checked by reading, not compiled, not run on hardware
//
#include <hip/hip_runtime.h>
#include <hip/hip_bf16.h>

// ---- problem constants (from reference) ----
#define LYRS     2
#define H_DIM    1024
#define I_DIM    2048
#define NHEADS   32
#define P_DIM    64
#define N_DIM    64
#define K_CONV   4
#define CONV_DIM 2176           // I + 2*G*N
#define PROJ_DIM 4256           // I + CONV_DIM + NH
#define PROJ_PAD 4288           // round up to 64 for block tiling (67*64)
#define SEQLEN   512
#define EPS_F    1e-5f
#define ROW_PAD  40             // bf16 elems per LDS tile row (32 data + 8 pad = 80B, 20 banks)

typedef __bf16 bf16_t;
typedef __attribute__((ext_vector_type(8)))  __bf16 v8bf;
typedef __attribute__((ext_vector_type(16))) __bf16 v16bf;
typedef __attribute__((ext_vector_type(8)))  float  v8f;
typedef int v4i_ __attribute__((vector_size(16)));   // matches builtin's V4i param type

// ---- CDNA5 async global->LDS path (ASYNCcnt), with safe fallback ----
#if defined(__HIP_DEVICE_COMPILE__) && __has_builtin(__builtin_amdgcn_global_load_async_to_lds_b128)
#define HAVE_ASYNC_LDS 1
#else
#define HAVE_ASYNC_LDS 0
#endif

#if HAVE_ASYNC_LDS
#define ASYNC_WAIT_2() asm volatile("s_wait_asynccnt 2" ::: "memory")
#define ASYNC_WAIT_0() asm volatile("s_wait_asynccnt 0" ::: "memory")
#else
#define ASYNC_WAIT_2()
#define ASYNC_WAIT_0()
#endif

__device__ __forceinline__ void async_copy16(const bf16_t* g, bf16_t* l) {
#if HAVE_ASYNC_LDS
    __builtin_amdgcn_global_load_async_to_lds_b128(
        (__attribute__((address_space(1))) v4i_*)g,
        (__attribute__((address_space(3))) v4i_*)l, 0, 0);
#else
    *(v8bf*)l = *(const v8bf*)g;    // sync fallback: global load + ds store
#endif
}

__device__ __forceinline__ v16bf mk16(v8bf lo, v8bf hi) {
    return __builtin_shufflevector(lo, hi, 0,1,2,3,4,5,6,7,8,9,10,11,12,13,14,15);
}

// ---------------- helpers ----------------
__device__ __forceinline__ float siluf(float x) { return x / (1.f + expf(-x)); }
__device__ __forceinline__ float softplusf(float x) { return (x > 20.f) ? x : log1pf(expf(x)); }

__device__ float block_reduce_sum_256(float v) {
    #pragma unroll
    for (int o = 16; o > 0; o >>= 1) v += __shfl_xor(v, o, 32);
    __shared__ float smem[8];
    __shared__ float total;
    const int lane = threadIdx.x & 31, w = threadIdx.x >> 5;
    if (lane == 0) smem[w] = v;
    __syncthreads();
    if (threadIdx.x == 0) {
        float t = 0.f;
        #pragma unroll
        for (int i = 0; i < 8; ++i) t += smem[i];
        total = t;
    }
    __syncthreads();
    return total;
}

// ---------------- small elementwise kernels ----------------
// copies n_valid elems as bf16, zero-fills up to n_total (trailing pad rows)
__global__ void convert_pad_bf16_kernel(const float* __restrict__ src,
                                        bf16_t* __restrict__ dst,
                                        int n_valid, int n_total) {
    int i = blockIdx.x * blockDim.x + threadIdx.x;
    if (i < n_total) dst[i] = (i < n_valid) ? (bf16_t)src[i] : (bf16_t)0.f;
}

// one block (256 thr) per row of 1024; out = bf16(rms(x) * w)
__global__ void rmsnorm_bf16_kernel(const float* __restrict__ x,
                                    const float* __restrict__ w,
                                    bf16_t* __restrict__ out) {
    const int s = blockIdx.x;
    const float* xr = x + (size_t)s * H_DIM;
    float v[4]; float ss = 0.f;
    #pragma unroll
    for (int i = 0; i < 4; ++i) { v[i] = xr[threadIdx.x + i * 256]; ss += v[i] * v[i]; }
    ss = block_reduce_sum_256(ss);
    const float sc = rsqrtf(ss * (1.f / H_DIM) + EPS_F);
    #pragma unroll
    for (int i = 0; i < 4; ++i) {
        const int c = threadIdx.x + i * 256;
        out[(size_t)s * H_DIM + c] = (bf16_t)(v[i] * sc * w[c]);
    }
}

// final norm: f32 out
__global__ void rmsnorm_f32_kernel(const float* __restrict__ x,
                                   const float* __restrict__ w,
                                   float* __restrict__ out) {
    const int s = blockIdx.x;
    const float* xr = x + (size_t)s * H_DIM;
    float v[4]; float ss = 0.f;
    #pragma unroll
    for (int i = 0; i < 4; ++i) { v[i] = xr[threadIdx.x + i * 256]; ss += v[i] * v[i]; }
    ss = block_reduce_sum_256(ss);
    const float sc = rsqrtf(ss * (1.f / H_DIM) + EPS_F);
    #pragma unroll
    for (int i = 0; i < 4; ++i) {
        const int c = threadIdx.x + i * 256;
        out[(size_t)s * H_DIM + c] = v[i] * sc * w[c];
    }
}

// ---------------- bf16 WMMA GEMM through LDS:  C[M,N] (+)= A[M,K] * B[N,K]^T ----------------
// Block tile 64(M) x 64(N), 8 waves as 4(M) x 2(N); each wave: two 16x16 WMMA accumulators.
// K-step 32. A/B tiles (4 KB each) double-buffered in LDS, filled with async global->LDS
// loads tracked by ASYNCcnt; fragments read back via ds_load_b128.
// B rows must be readable up to gridDim.x*64 rows (caller pads weights).
__global__ void __launch_bounds__(256)
gemm_bf16_wmma_lds_kernel(const bf16_t* __restrict__ A,
                          const bf16_t* __restrict__ B,
                          float* __restrict__ C,
                          int N, int K, int accumulate) {
    __shared__ bf16_t As[2][64 * ROW_PAD];
    __shared__ bf16_t Bs[2][64 * ROW_PAD];

    const int tid  = threadIdx.x;
    const int lane = tid & 31;
    const int wv   = tid >> 5;
    const int mw   = wv & 3;            // M sub-tile 0..3
    const int nw   = wv >> 2;           // N sub-tile 0..1
    const int mBase = blockIdx.y * 64;
    const int nBase = blockIdx.x * 64;
    const int half  = (lane >> 4) & 1;  // 0: lanes 0-15, 1: lanes 16-31
    const int r16   = lane & 15;

    // staging: thread t fills 16B chunk (t&3) of tile row (t>>2)
    const int ldRow = tid >> 2;
    const int ldChk = (tid & 3) * 8;    // element offset within 32-elem row
    const bf16_t* gA = A + (size_t)(mBase + ldRow) * K + ldChk;
    const bf16_t* gB = B + (size_t)(nBase + ldRow) * K + ldChk;
    const int ldsOff = ldRow * ROW_PAD + ldChk;

    const int S = K >> 5;               // K-steps of 32
    // prologue: stage steps 0 and 1 (K >= 64 always here)
    async_copy16(gA,      &As[0][ldsOff]);
    async_copy16(gB,      &Bs[0][ldsOff]);
    async_copy16(gA + 32, &As[1][ldsOff]);
    async_copy16(gB + 32, &Bs[1][ldsOff]);

    v8f acc0 = {0.f,0.f,0.f,0.f,0.f,0.f,0.f,0.f};
    v8f acc1 = {0.f,0.f,0.f,0.f,0.f,0.f,0.f,0.f};
    // fragment LDS offsets (contiguous k order, same for A and B):
    // A lane: row = mw*16 + lane%16, K chunk {0..15} / {16..31}
    // B lane: col-row = nw*32 (+16) + lane%16, K chunk {0..15} / {16..31}
    const int aIdx  = (mw * 16 + r16) * ROW_PAD + half * 16;
    const int bIdx0 = (nw * 32 + r16) * ROW_PAD + half * 16;
    const int bIdx1 = bIdx0 + 16 * ROW_PAD;

    for (int s = 0; s < S; ++s) {
        const int buf = s & 1;
        if (s < S - 1) { ASYNC_WAIT_2(); } else { ASYNC_WAIT_0(); }
        __syncthreads();                 // tile for step s visible to all waves
        v16bf av  = mk16(*(const v8bf*)&As[buf][aIdx],      *(const v8bf*)&As[buf][aIdx + 8]);
        v16bf bv0 = mk16(*(const v8bf*)&Bs[buf][bIdx0],     *(const v8bf*)&Bs[buf][bIdx0 + 8]);
        v16bf bv1 = mk16(*(const v8bf*)&Bs[buf][bIdx1],     *(const v8bf*)&Bs[buf][bIdx1 + 8]);
        acc0 = __builtin_amdgcn_wmma_f32_16x16x32_bf16(false, av, false, bv0, (short)0, acc0, false, false);
        acc1 = __builtin_amdgcn_wmma_f32_16x16x32_bf16(false, av, false, bv1, (short)0, acc1, false, false);
        __syncthreads();                 // all waves done reading buf before refill
        if (s + 2 < S) {
            const int k0 = (s + 2) * 32;
            async_copy16(gA + k0, &As[buf][ldsOff]);
            async_copy16(gB + k0, &Bs[buf][ldsOff]);
        }
    }

    // C/D layout: VGPR r -> M = r + 8*half, N = lane%16
    const int m0 = mBase + mw * 16 + half * 8;
    const int n0 = nBase + nw * 32 + r16;
    #pragma unroll
    for (int r = 0; r < 8; ++r) {
        if (n0 < N) {
            float* p = C + (size_t)(m0 + r) * N + n0;
            float v = acc0[r]; if (accumulate) v += *p; *p = v;
        }
        if (n0 + 16 < N) {
            float* p = C + (size_t)(m0 + r) * N + n0 + 16;
            float v = acc1[r]; if (accumulate) v += *p; *p = v;
        }
    }
}

// ---------------- depthwise causal conv (K=4) + SiLU ----------------
__global__ void conv_silu_kernel(const float* __restrict__ proj,
                                 const float* __restrict__ cw,
                                 const float* __restrict__ cb,
                                 float* __restrict__ hBC) {
    const int idx = blockIdx.x * blockDim.x + threadIdx.x;
    if (idx >= SEQLEN * CONV_DIM) return;
    const int s = idx / CONV_DIM;
    const int c = idx - s * CONV_DIM;
    float v = cb[c];
    #pragma unroll
    for (int k = 0; k < K_CONV; ++k) {
        const int t = s - (K_CONV - 1) + k;
        if (t >= 0) v += proj[(size_t)t * PROJ_DIM + I_DIM + c] * cw[c * K_CONV + k];
    }
    hBC[(size_t)s * CONV_DIM + c] = siluf(v);
}

// ---------------- sequential SSM scan ----------------
__global__ void ssm_scan_kernel(const float* __restrict__ proj,
                                const float* __restrict__ hBC,
                                const float* __restrict__ A_log,
                                const float* __restrict__ dt_bias,
                                const float* __restrict__ Dp,
                                float* __restrict__ y) {
    const int tid = threadIdx.x;
    const int h   = blockIdx.x * 4 + (tid >> 6);
    const int p   = tid & 63;
    __shared__ float sB[N_DIM];
    __shared__ float sC[N_DIM];

    float state[N_DIM];
    #pragma unroll
    for (int n = 0; n < N_DIM; ++n) state[n] = 0.f;

    const float Aneg = -expf(A_log[h]);
    const float dtb  = dt_bias[h];
    const float Dh   = Dp[h];

    for (int s = 0; s < SEQLEN; ++s) {
        if (tid < 64)        sB[tid]      = hBC[(size_t)s * CONV_DIM + I_DIM + tid];
        else if (tid < 128)  sC[tid - 64] = hBC[(size_t)s * CONV_DIM + I_DIM + N_DIM + (tid - 64)];
        __syncthreads();

        float dtv = softplusf(proj[(size_t)s * PROJ_DIM + (I_DIM + CONV_DIM) + h] + dtb);
        const float dA   = expf(dtv * Aneg);
        const float x    = hBC[(size_t)s * CONV_DIM + h * P_DIM + p];
        const float coef = dtv * x;
        float acc = 0.f;
        #pragma unroll
        for (int n = 0; n < N_DIM; ++n) {
            state[n] = dA * state[n] + coef * sB[n];
            acc += state[n] * sC[n];
        }
        y[(size_t)s * I_DIM + h * P_DIM + p] = acc + Dh * x;
        __syncthreads();
    }
}

// ---------------- gate (silu) + RMSNorm over I, emit bf16 for out_proj ----------------
__global__ void gate_norm_bf16_kernel(const float* __restrict__ y,
                                      const float* __restrict__ proj,
                                      const float* __restrict__ gw,
                                      bf16_t* __restrict__ out) {
    const int s = blockIdx.x;
    float z[8]; float ss = 0.f;
    #pragma unroll
    for (int i = 0; i < 8; ++i) {
        const int c = threadIdx.x + i * 256;
        const float g = proj[(size_t)s * PROJ_DIM + c];   // gate = proj[..., :I]
        const float v = y[(size_t)s * I_DIM + c] * siluf(g);
        z[i] = v; ss += v * v;
    }
    ss = block_reduce_sum_256(ss);
    const float sc = rsqrtf(ss * (1.f / I_DIM) + EPS_F);
    #pragma unroll
    for (int i = 0; i < 8; ++i) {
        const int c = threadIdx.x + i * 256;
        out[(size_t)s * I_DIM + c] = (bf16_t)(z[i] * sc * gw[c]);
    }
}

// ---------------- host orchestration ----------------
extern "C" void kernel_launch(void* const* d_in, const int* in_sizes, int n_in,
                              void* d_out, int out_size, void* d_ws, size_t ws_size,
                              hipStream_t stream) {
    (void)in_sizes; (void)n_in; (void)out_size; (void)ws_size;

    const float* hidden      = (const float*)d_in[0];
    const float* norm_w      = (const float*)d_in[1];
    const float* in_proj_w   = (const float*)d_in[2];
    const float* conv_w      = (const float*)d_in[3];
    const float* conv_b      = (const float*)d_in[4];
    const float* A_log       = (const float*)d_in[5];
    const float* dt_bias     = (const float*)d_in[6];
    const float* Dp          = (const float*)d_in[7];
    const float* gate_norm_w = (const float*)d_in[8];
    const float* out_proj_w  = (const float*)d_in[9];
    const float* norm_f_w    = (const float*)d_in[10];
    float* out = (float*)d_out;

    // workspace carve-up (all chunk sizes multiples of 256B), ~34 MB total
    char* base = (char*)d_ws;
    size_t o = 0;
    float*  h_buf  = (float*)(base + o);  o += (size_t)SEQLEN * H_DIM    * 4;  // residual stream
    bf16_t* hn_bf  = (bf16_t*)(base + o); o += (size_t)SEQLEN * H_DIM    * 2;  // rmsnorm out
    bf16_t* wb_in  = (bf16_t*)(base + o); o += (size_t)PROJ_PAD * H_DIM  * 2;  // in_proj bf16 (padded rows)
    bf16_t* wb_out = (bf16_t*)(base + o); o += (size_t)H_DIM * I_DIM     * 2;  // out_proj bf16
    float*  proj   = (float*)(base + o);  o += (size_t)SEQLEN * PROJ_DIM * 4;
    float*  hBC    = (float*)(base + o);  o += (size_t)SEQLEN * CONV_DIM * 4;
    float*  yscan  = (float*)(base + o);  o += (size_t)SEQLEN * I_DIM    * 4;
    bf16_t* ybf    = (bf16_t*)(base + o); o += (size_t)SEQLEN * I_DIM    * 2;

    (void)hipMemcpyAsync(h_buf, hidden, (size_t)SEQLEN * H_DIM * 4,
                         hipMemcpyDeviceToDevice, stream);

    const dim3 blk(256);
    const dim3 gGemm1(PROJ_PAD / 64, SEQLEN / 64);   // (67, 8)
    const dim3 gGemm2(H_DIM / 64,    SEQLEN / 64);   // (16, 8)
    const int  nConv = SEQLEN * CONV_DIM;

    for (int l = 0; l < LYRS; ++l) {
        const float* w_in_l  = in_proj_w  + (size_t)l * PROJ_DIM * H_DIM;
        const float* w_out_l = out_proj_w + (size_t)l * H_DIM * I_DIM;

        // 1) RMSNorm -> bf16 activations
        rmsnorm_bf16_kernel<<<SEQLEN, blk, 0, stream>>>(h_buf, norm_w + l * H_DIM, hn_bf);

        // 2) weights -> bf16 (in_proj zero-padded to 4288 rows for 64-wide N tiling)
        convert_pad_bf16_kernel<<<(PROJ_PAD * H_DIM + 255) / 256, blk, 0, stream>>>(
            w_in_l, wb_in, PROJ_DIM * H_DIM, PROJ_PAD * H_DIM);
        convert_pad_bf16_kernel<<<(H_DIM * I_DIM + 255) / 256, blk, 0, stream>>>(
            w_out_l, wb_out, H_DIM * I_DIM, H_DIM * I_DIM);

        // 3) in_proj GEMM: WMMA + async-LDS double buffering
        gemm_bf16_wmma_lds_kernel<<<gGemm1, blk, 0, stream>>>(
            hn_bf, wb_in, proj, PROJ_DIM, H_DIM, 0);

        // 4) causal depthwise conv + SiLU
        conv_silu_kernel<<<(nConv + 255) / 256, blk, 0, stream>>>(
            proj, conv_w + (size_t)l * CONV_DIM * K_CONV, conv_b + (size_t)l * CONV_DIM, hBC);

        // 5) sequential selective-state scan (states in VGPRs, B/C via LDS)
        ssm_scan_kernel<<<NHEADS / 4, blk, 0, stream>>>(
            proj, hBC, A_log + l * NHEADS, dt_bias + l * NHEADS, Dp + l * NHEADS, yscan);

        // 6) gate + RMSNorm -> bf16
        gate_norm_bf16_kernel<<<SEQLEN, blk, 0, stream>>>(
            yscan, proj, gate_norm_w + (size_t)l * I_DIM, ybf);

        // 7) out_proj GEMM on WMMA, fused residual add into h_buf
        gemm_bf16_wmma_lds_kernel<<<gGemm2, blk, 0, stream>>>(
            ybf, wb_out, h_buf, H_DIM, I_DIM, 1);
    }

    // final norm -> f32 output
    rmsnorm_f32_kernel<<<SEQLEN, blk, 0, stream>>>(h_buf, norm_f_w, out);
}
